// Decoder_9586367004779
// MI455X (gfx1250) — matrix operand order repaired and checked by reading, compiled, not run on hardware
//
#include <hip/hip_runtime.h>
#include <hip/hip_bf16.h>

// ---------------- problem constants ----------------
#define S_    16
#define P_    32
#define H_    128
#define E_    64
#define BN_   1024
#define MH_   1024
#define T_    8
#define B_    (S_ * P_)        // 512
#define RPOOL (S_ * P_ * P_)   // 16384 pairwise rows
#define KPOOL (E_ + H_)        // 192
#define KMLP  (H_ + BN_)       // 1152

typedef __attribute__((ext_vector_type(16))) __bf16 v16bf;
typedef __attribute__((ext_vector_type(8)))  float  v8f;

union FragBf { v16bf v; unsigned short u[16]; uint4 q[2]; };

// fp32 -> bf16 round-to-nearest-even
__device__ __forceinline__ unsigned short f2bf(float f) {
  unsigned int u = __float_as_uint(f);
  u += 0x7fffu + ((u >> 16) & 1u);
  return (unsigned short)(u >> 16);
}

__device__ __forceinline__ float sigmoidf_(float x) {
  return 1.0f / (1.0f + __expf(-x));
}

// deterministic counter-based uniform in (0,1)
__device__ __forceinline__ float hash_uniform(unsigned int x) {
  x ^= x >> 16; x *= 0x7feb352du;
  x ^= x >> 15; x *= 0x846ca68bu;
  x ^= x >> 16;
  return ((x >> 8) + 0.5f) * (1.0f / 16777216.0f);
}

// ---------------- fp32 [K,N] -> bf16 transposed [N,K] ----------------
__global__ void k_cvt_bf16_T(const float* __restrict__ src,
                             unsigned short* __restrict__ dst, int K, int N) {
  const int total = K * N;
  for (int i = blockIdx.x * blockDim.x + threadIdx.x; i < total;
       i += gridDim.x * blockDim.x) {
    const int n = i / K, k = i - n * K;
    dst[i] = f2bf(src[(size_t)k * N + n]);   // dst[n*K+k] = src[k*N+n]
  }
}

// ---------------- LSTM cell + position head ----------------
// one block (128 threads) per ped row b
__global__ __launch_bounds__(128)
void k_lstm(const float* __restrict__ W_se, const float* __restrict__ b_se,
            const float* __restrict__ W_ih, const float* __restrict__ W_hh,
            const float* __restrict__ b_ih, const float* __restrict__ b_hh,
            const float* __restrict__ W_hp, const float* __restrict__ b_hp,
            const float* __restrict__ h_in, float* __restrict__ c_st,
            float* __restrict__ h_lstm, float* __restrict__ pos,
            float* __restrict__ lr, float* __restrict__ out, int t) {
  __shared__ float xe[E_], hs[H_], hn[H_];
  const int b = blockIdx.x, n = threadIdx.x;
  const float l0 = lr[b * 2 + 0], l1 = lr[b * 2 + 1];
  if (n < E_) xe[n] = l0 * W_se[n] + l1 * W_se[E_ + n] + b_se[n];
  hs[n] = h_in[b * H_ + n];
  __syncthreads();

  float ai = b_ih[n]        + b_hh[n];
  float af = b_ih[128 + n]  + b_hh[128 + n];
  float ag = b_ih[256 + n]  + b_hh[256 + n];
  float ao = b_ih[384 + n]  + b_hh[384 + n];
  for (int k = 0; k < E_; ++k) {
    const float x = xe[k]; const float* w = &W_ih[k * 512];
    ai += x * w[n]; af += x * w[128 + n]; ag += x * w[256 + n]; ao += x * w[384 + n];
  }
  for (int k = 0; k < H_; ++k) {
    const float x = hs[k]; const float* w = &W_hh[k * 512];
    ai += x * w[n]; af += x * w[128 + n]; ag += x * w[256 + n]; ao += x * w[384 + n];
  }
  const float cn = sigmoidf_(af) * c_st[b * H_ + n] + sigmoidf_(ai) * tanhf(ag);
  const float hv = sigmoidf_(ao) * tanhf(cn);
  c_st[b * H_ + n] = cn;
  h_lstm[b * H_ + n] = hv;
  hn[n] = hv;
  __syncthreads();

  if (n < 2) {
    float r = b_hp[n];
    for (int k = 0; k < H_; ++k) r += hn[k] * W_hp[k * 2 + n];
    out[t * (B_ * 2) + b * 2 + n] = r;          // pred_traj_fake_rel
    pos[b * 2 + n] += r;                        // curr_pos
    lr[b * 2 + n] = r;                          // next step's rel input
  }
}

// ---------------- build pool A matrix [16384,192] bf16 ----------------
__global__ void k_build_A_pool(const float* __restrict__ pos,
                               const float* __restrict__ h_lstm,
                               const float* __restrict__ W_pse,
                               const float* __restrict__ b_pse,
                               unsigned short* __restrict__ A) {
  const int total = RPOOL * KPOOL;
  for (int idx = blockIdx.x * blockDim.x + threadIdx.x; idx < total;
       idx += gridDim.x * blockDim.x) {
    const int r = idx / KPOOL, k = idx - r * KPOOL;
    const int s = r >> 10, i = (r >> 5) & 31, j = r & 31;
    float val;
    if (k < E_) {
      const int bi = s * P_ + i, bj = s * P_ + j;
      const float dx = pos[bj * 2 + 0] - pos[bi * 2 + 0];
      const float dy = pos[bj * 2 + 1] - pos[bi * 2 + 1];
      val = dx * W_pse[k] + dy * W_pse[E_ + k] + b_pse[k];
    } else {
      val = h_lstm[(s * P_ + j) * H_ + (k - E_)];
    }
    A[idx] = f2bf(val);
  }
}

// ---------------- build MLP A matrix [512,1152] bf16 ----------------
__global__ void k_build_A_mlp(const float* __restrict__ h_lstm,
                              const float* __restrict__ pooled,
                              unsigned short* __restrict__ A) {
  const int total = B_ * KMLP;
  for (int idx = blockIdx.x * blockDim.x + threadIdx.x; idx < total;
       idx += gridDim.x * blockDim.x) {
    const int r = idx / KMLP, k = idx - r * KMLP;
    const float val = (k < H_) ? h_lstm[r * H_ + k] : pooled[r * BN_ + (k - H_)];
    A[idx] = f2bf(val);
  }
}

// -------- fragment builders: contiguous 16B runs -> ds_load_b128 ----------
// A tile As[64][32] row-major; lane fragment = rows of K: {8kh..8kh+7, 16+8kh..+7}
__device__ __forceinline__ void load_a_frag(const unsigned short* __restrict__ As,
                                            int mrow, int khalf, FragBf& a) {
  const unsigned short* ap = &As[mrow * 32 + 8 * khalf];
  a.q[0] = *(const uint4*)ap;          // k = 8kh .. 8kh+7
  a.q[1] = *(const uint4*)(ap + 16);   // k = 16+8kh .. 16+8kh+7
}
// B tile staged TRANSPOSED: BsT[64][32] (n-major). Lane fragment = column n,
// k = {8kh..8kh+7, 16+8kh..16+8kh+7} (matches A fragment).
__device__ __forceinline__ void load_b_frag(const unsigned short* __restrict__ BsT,
                                            int ncol, int khalf, FragBf& b) {
  const unsigned short* bp = &BsT[ncol * 32 + 8 * khalf];
  b.q[0] = *(const uint4*)bp;
  b.q[1] = *(const uint4*)(bp + 16);
}

// ---------------- generic bf16 WMMA GEMM: out = relu(A @ B + bias) --------
// A row-major [M,K] bf16; BmT TRANSPOSED weights [N,K] bf16.
// Tile 64x64, 4 waves (128 thr). Statically-scheduled double buffer.
// Order per chunk: compute(cur) issues WMMAs, then gload+sstore(next) runs
// while WMMAs execute -> staged regs live only load->store, no spills.
__global__ __launch_bounds__(128, 1)
void k_gemm_bf16_relu(const unsigned short* __restrict__ A,
                      const unsigned short* __restrict__ BmT,
                      const float* __restrict__ bias,
                      unsigned short* __restrict__ outB,
                      float* __restrict__ outF,
                      int M, int N, int K) {
  __shared__ __align__(16) unsigned short As[2][64 * 32];
  __shared__ __align__(16) unsigned short BsT[2][64 * 32];
  const int tid = threadIdx.x, lane = tid & 31, wave = tid >> 5;
  const int lid = lane & 15, khalf = lane >> 4;
  const int m0 = blockIdx.y * 64, n0 = blockIdx.x * 64;

  v8f acc[4] = {v8f{}, v8f{}, v8f{}, v8f{}};

  // staging map: tile = 256 uint4 (64 rows x 4 uint4), thread -> 2 uint4
  auto stage = [&](int kk, int buf) {           // global -> regs -> LDS[buf]
    uint4 ar[2], br[2];
    #pragma unroll
    for (int i = 0; i < 2; ++i) {
      const int e = tid + i * 128;
      const int rr = e >> 2, cc = (e & 3) << 3;
      ar[i] = *(const uint4*)&A[(size_t)(m0 + rr) * K + kk + cc];
      br[i] = *(const uint4*)&BmT[(size_t)(n0 + rr) * K + kk + cc];
    }
    #pragma unroll
    for (int i = 0; i < 2; ++i) {
      const int e = tid + i * 128;
      const int rr = e >> 2, cc = (e & 3) << 3;
      *(uint4*)&As[buf][rr * 32 + cc]  = ar[i];
      *(uint4*)&BsT[buf][rr * 32 + cc] = br[i];
    }
  };
  auto compute = [&](int buf) {
    FragBf a;
    load_a_frag(&As[buf][0], wave * 16 + lid, khalf, a);
    #pragma unroll
    for (int nt = 0; nt < 4; ++nt) {
      FragBf b;
      load_b_frag(&BsT[buf][0], nt * 16 + lid, khalf, b);
      acc[nt] = __builtin_amdgcn_wmma_f32_16x16x32_bf16(
          false, a.v, false, b.v, (short)0, acc[nt], false, false);
    }
  };

  // chunk c lives in buffer (c & 1); chunk count K/32 is even in all cases
  stage(0, 0); __syncthreads();
  #pragma unroll 1
  for (int kk = 32; kk < K - 32; kk += 64) {
    compute(0); stage(kk, 1);      __syncthreads();
    compute(1); stage(kk + 32, 0); __syncthreads();
  }
  compute(0); stage(K - 32, 1); __syncthreads();
  compute(1);

  #pragma unroll
  for (int nt = 0; nt < 4; ++nt) {
    const int col = n0 + nt * 16 + lid;
    const float bs = bias[col];
    #pragma unroll
    for (int v = 0; v < 8; ++v) {                 // C layout: M=v+8*khalf, N=lid
      const int row = m0 + wave * 16 + v + 8 * khalf;
      float val = acc[nt][v] + bs;
      val = val > 0.f ? val : 0.f;
      if (outB) outB[(size_t)row * N + col] = f2bf(val);
      if (outF) outF[(size_t)row * N + col] = val;
    }
  }
}

// ------- fused pool GEMM2 + Gumbel hard selection over the j axis ---------
// Block = one (s,i) group: 32 rows x 64 cols, 8 waves (256 thr).
// Statically-scheduled double buffer; raw never touches HBM;
// pooled[b, d] = raw[argmax_j(log(raw+eps)+gumbel)].
__global__ __launch_bounds__(256, 1)
void k_pool_gemm2_gumbel(const unsigned short* __restrict__ X1,    // [16384,512] bf16
                         const unsigned short* __restrict__ Wp2T,  // [1024,512] bf16 (transposed)
                         const float* __restrict__ bias,           // [1024]
                         float* __restrict__ pooled,               // [512,1024]
                         int t) {
  __shared__ __align__(16) unsigned short As[2][32 * 32];
  __shared__ __align__(16) unsigned short BsT[2][64 * 32];
  __shared__ float rawS[32 * 64];
  const int tid = threadIdx.x, lane = tid & 31, wave = tid >> 5;
  const int lid = lane & 15, khalf = lane >> 4;
  const int g = blockIdx.y;                 // (s,i) flattened == ped row b
  const int n0 = blockIdx.x * 64;
  const int mstrip = wave & 1, ntile = wave >> 1;
  const bool hasA = tid < 128;              // A tile = 128 uint4

  v8f acc = {};

  auto stage = [&](int kk, int buf) {
    const int rr = tid >> 2, cc = (tid & 3) << 3;
    uint4 ar, br;
    if (hasA)
      ar = *(const uint4*)&X1[(size_t)(g * 32 + rr) * 512 + kk + cc];
    br = *(const uint4*)&Wp2T[(size_t)(n0 + rr) * 512 + kk + cc];
    if (hasA) *(uint4*)&As[buf][rr * 32 + cc] = ar;
    *(uint4*)&BsT[buf][rr * 32 + cc] = br;
  };
  auto compute = [&](int buf) {
    FragBf a, b;
    load_a_frag(&As[buf][0], mstrip * 16 + lid, khalf, a);
    load_b_frag(&BsT[buf][0], ntile * 16 + lid, khalf, b);
    acc = __builtin_amdgcn_wmma_f32_16x16x32_bf16(
        false, a.v, false, b.v, (short)0, acc, false, false);
  };

  stage(0, 0); __syncthreads();
  #pragma unroll 1
  for (int kk = 32; kk < 512 - 32; kk += 64) {
    compute(0); stage(kk, 1);      __syncthreads();
    compute(1); stage(kk + 32, 0); __syncthreads();
  }
  compute(0); stage(512 - 32, 1); __syncthreads();
  compute(1);

  #pragma unroll
  for (int v = 0; v < 8; ++v) {
    const int row = mstrip * 16 + v + 8 * khalf;
    const int col = ntile * 16 + lid;
    const float val = acc[v] + bias[n0 + col];
    rawS[row * 64 + col] = val > 0.f ? val : 0.f;      // relu -> raw
  }
  __syncthreads();

  if (tid < 64) {                                      // column-wise gumbel argmax
    const int d = n0 + tid;
    float best = -3.4e38f; int bj = 0;
    for (int j = 0; j < 32; ++j) {
      const float r = rawS[j * 64 + tid];
      const unsigned int idx =
          (((unsigned)t * 512u + (unsigned)g) * 32u + (unsigned)j) * 1024u +
          (unsigned)d;
      const float u = hash_uniform(idx ^ 0x85ebca6bu);
      const float gum = -logf(1e-10f - logf(u + 1e-10f));
      const float z = logf(r + 1e-8f) + gum;
      if (z > best) { best = z; bj = j; }
    }
    pooled[(size_t)g * BN_ + d] = rawS[bj * 64 + tid]; // hard straight-through
  }
}

// ---------------------------- host launcher -------------------------------
extern "C" void kernel_launch(void* const* d_in, const int* in_sizes, int n_in,
                              void* d_out, int out_size, void* d_ws, size_t ws_size,
                              hipStream_t stream) {
  const float* last_pos = (const float*)d_in[0];
  const float* last_rel = (const float*)d_in[1];
  const float* h0   = (const float*)d_in[2];
  const float* c0   = (const float*)d_in[3];
  const float* W_se = (const float*)d_in[4];  const float* b_se = (const float*)d_in[5];
  const float* W_ih = (const float*)d_in[6];  const float* W_hh = (const float*)d_in[7];
  const float* b_ih = (const float*)d_in[8];  const float* b_hh = (const float*)d_in[9];
  const float* W_hp = (const float*)d_in[10]; const float* b_hp = (const float*)d_in[11];
  const float* W_pse= (const float*)d_in[12]; const float* b_pse= (const float*)d_in[13];
  const float* W_p1 = (const float*)d_in[14]; const float* b_p1 = (const float*)d_in[15];
  const float* W_p2 = (const float*)d_in[16]; const float* b_p2 = (const float*)d_in[17];
  const float* W_m1 = (const float*)d_in[18]; const float* b_m1 = (const float*)d_in[19];
  const float* W_m2 = (const float*)d_in[20]; const float* b_m2 = (const float*)d_in[21];
  float* out = (float*)d_out;

  // workspace carve-up (256B aligned)
  char* base = (char*)d_ws;
  size_t off = 0;
  auto alloc = [&](size_t bytes) -> void* {
    void* p = base + off;
    off = (off + bytes + 255) & ~(size_t)255;
    return p;
  };
  unsigned short* Wp1T = (unsigned short*)alloc((size_t)KPOOL * 512 * 2);  // [512,192]
  unsigned short* Wp2T = (unsigned short*)alloc((size_t)512 * BN_ * 2);    // [1024,512]
  unsigned short* Wm1T = (unsigned short*)alloc((size_t)KMLP * MH_ * 2);   // [1024,1152]
  unsigned short* Wm2T = (unsigned short*)alloc((size_t)MH_ * H_ * 2);     // [128,1024]
  float* h_state = (float*)alloc((size_t)B_ * H_ * 4);
  float* c_state = (float*)alloc((size_t)B_ * H_ * 4);
  float* h_lstm  = (float*)alloc((size_t)B_ * H_ * 4);
  float* pos     = (float*)alloc((size_t)B_ * 2 * 4);
  float* lr      = (float*)alloc((size_t)B_ * 2 * 4);
  float* pooled  = (float*)alloc((size_t)B_ * BN_ * 4);
  unsigned short* A_pool = (unsigned short*)alloc((size_t)RPOOL * KPOOL * 2);
  unsigned short* X1     = (unsigned short*)alloc((size_t)RPOOL * 512 * 2);
  unsigned short* A_mlp  = (unsigned short*)alloc((size_t)B_ * KMLP * 2);
  unsigned short* X_mlp  = (unsigned short*)alloc((size_t)B_ * MH_ * 2);

  // constant weights -> transposed bf16 (once per call; deterministic)
  auto cvtT = [&](const float* s, unsigned short* d, int K, int N) {
    k_cvt_bf16_T<<<(K * N + 255) / 256, 256, 0, stream>>>(s, d, K, N);
  };
  cvtT(W_p1, Wp1T, KPOOL, 512);
  cvtT(W_p2, Wp2T, 512, BN_);
  cvtT(W_m1, Wm1T, KMLP, MH_);
  cvtT(W_m2, Wm2T, MH_, H_);

  // seed recurrent state (d2d async copies are graph-capture safe)
  hipMemcpyAsync(h_state, h0,       (size_t)B_ * H_ * 4, hipMemcpyDeviceToDevice, stream);
  hipMemcpyAsync(c_state, c0,       (size_t)B_ * H_ * 4, hipMemcpyDeviceToDevice, stream);
  hipMemcpyAsync(pos,     last_pos, (size_t)B_ * 2 * 4,  hipMemcpyDeviceToDevice, stream);
  hipMemcpyAsync(lr,      last_rel, (size_t)B_ * 2 * 4,  hipMemcpyDeviceToDevice, stream);

  for (int t = 0; t < T_; ++t) {
    k_lstm<<<B_, 128, 0, stream>>>(W_se, b_se, W_ih, W_hh, b_ih, b_hh, W_hp, b_hp,
                                   h_state, c_state, h_lstm, pos, lr, out, t);

    k_build_A_pool<<<1024, 256, 0, stream>>>(pos, h_lstm, W_pse, b_pse, A_pool);

    // X1 = relu(A_pool[16384,192] @ W_p1[192,512] + b_p1)
    k_gemm_bf16_relu<<<dim3(512 / 64, RPOOL / 64), 128, 0, stream>>>(
        A_pool, Wp1T, b_p1, X1, nullptr, RPOOL, 512, KPOOL);

    // pooled = gumbel-hard-select over j of relu(X1 @ W_p2 + b_p2)
    k_pool_gemm2_gumbel<<<dim3(BN_ / 64, B_), 256, 0, stream>>>(
        X1, Wp2T, b_p2, pooled, t);

    k_build_A_mlp<<<512, 256, 0, stream>>>(h_lstm, pooled, A_mlp);

    // X_mlp = relu(A_mlp[512,1152] @ W_m1[1152,1024] + b_m1)
    k_gemm_bf16_relu<<<dim3(MH_ / 64, B_ / 64), 128, 0, stream>>>(
        A_mlp, Wm1T, b_m1, X_mlp, nullptr, B_, MH_, KMLP);

    // h_state = relu(X_mlp[512,1024] @ W_m2[1024,128] + b_m2)  (fp32 out)
    k_gemm_bf16_relu<<<dim3(H_ / 64, B_ / 64), 128, 0, stream>>>(
        X_mlp, Wm2T, b_m2, nullptr, h_state, B_, H_, MH_);
  }

  (void)in_sizes; (void)n_in; (void)out_size; (void)ws_size;
}
